// GATRootCauseRanker_42606075576890
// MI455X (gfx1250) — matrix operand order repaired and checked by reading, not compiled, not run on hardware
//
#include <hip/hip_runtime.h>
#include <hip/hip_bf16.h>
#include <cstdint>

typedef __attribute__((ext_vector_type(16))) _Float16 v16h;
typedef __attribute__((ext_vector_type(8)))  float    v8f;

#define HID 72
#define HEADS 4
#define NEG_SLOPE 0.2f

// Swizzle within a 32-k group so each half-wave's WMMA A-fragment is one
// contiguous 32B run: positions [k0..7, k16..23, k8..15, k24..31]
// (i.e. swap bits 3 and 4 of k%32).
__host__ __device__ __forceinline__ int aswz(int km) {
    return ((km & 8) << 1) | ((km & 16) >> 1) | (km & 7);
}

// ---------------------------------------------------------------------------
// WMMA GEMM: C[M x Nact](f32) = A[M x K](f16 swizzled, lda) * B(f16 packed, ldb)
// K multiple of 32, M multiple of 16. Each wave computes one 16x16 tile.
// A layout: A[row*lda + (k&~31) + aswz(k%32)]
// B layout: B[((k/32)*ldb + col)*32 + aswz(k%32)]
// act: 0=none, 1=elu, 2=relu. bias may be null.
// ---------------------------------------------------------------------------
__global__ __launch_bounds__(256)
void wmma_gemm_kernel(const _Float16* __restrict__ A, int lda,
                      const _Float16* __restrict__ B, int ldb,
                      float* __restrict__ C, int ldc,
                      int K, int Nact,
                      const float* __restrict__ bias, int act)
{
    const int lane = threadIdx.x & 31;
    const int wave = threadIdx.x >> 5;
    const int tn   = blockIdx.y * 8 + wave;
    const int n0   = tn * 16;
    if (n0 >= Nact) return;                 // wave-uniform exit (EXEC stays full)
    const int m0   = blockIdx.x * 16;
    const int l15  = lane & 15;
    const int hi   = (lane & 16) ? 1 : 0;   // half-wave selector

    // A: one contiguous 16-half (32B) run per lane per k-step
    const _Float16* arow = A + (size_t)(m0 + l15) * lda + hi * 16;
    // B: one contiguous 16-half (32B) run per lane per k-step
    const _Float16* bcol = B + ((size_t)(n0 + l15)) * 32 + hi * 16;

    v8f acc = {};
    for (int kk = 0; kk < K; kk += 32) {
        v16h a = *(const v16h*)(arow + kk);
        v16h b = *(const v16h*)(bcol + (size_t)(kk >> 5) * ldb * 32);
        acc = __builtin_amdgcn_wmma_f32_16x16x32_f16(
                  false, a, false, b, (short)0, acc, false, false);
    }

    // D layout: VGPR r -> row m0 + r + (lane>=16 ? 8 : 0), col = n0 + l15
    const int c = n0 + l15;
    if (c < Nact) {
        const float bv = bias ? bias[c] : 0.0f;
        float* crow = C + (size_t)(m0 + hi * 8) * ldc + c;
#pragma unroll
        for (int r = 0; r < 8; ++r) {
            float v = acc[r] + bv;
            if (act == 1)      v = (v > 0.0f) ? v : (__expf(v) - 1.0f);
            else if (act == 2) v = fmaxf(v, 0.0f);
            crow[(size_t)r * ldc] = v;
        }
    }
}

// f32 [M x n] -> f16 [M x ldp] in A-swizzled layout, zero pad cols [n, ldp)
__global__ void convert_pad_kernel(const float* __restrict__ in,
                                   _Float16* __restrict__ out,
                                   int n, int ldp, long long total)
{
    long long t = (long long)blockIdx.x * blockDim.x + threadIdx.x;
    if (t >= total) return;
    int j = (int)(t % ldp);                 // logical k index
    long long r = t / ldp;
    _Float16 v = (j < n) ? (_Float16)in[r * n + j] : (_Float16)0.0f;
    out[r * ldp + (j & ~31) + aswz(j & 31)] = v;
}

// Build packed f16 B matrix from f32 W [K x Nf], appending 2H attention-logit
// columns: col Nf+h = W_head_h @ a_src[h], col Nf+H+h = W_head_h @ a_dst[h].
// Output layout: out[((k/32)*ldo + j)*32 + aswz(k%32)], k < Kp (zero pad k >= K).
__global__ void prep_b_kernel(const float* __restrict__ W,
                              const float* __restrict__ as,
                              const float* __restrict__ ad,
                              int K, int Nf, int H, int C,
                              _Float16* __restrict__ out, int ldo, int total)
{
    int t = blockIdx.x * blockDim.x + threadIdx.x;
    if (t >= total) return;
    int j = t % ldo, k = t / ldo;
    float v = 0.0f;
    if (k < K) {
        if (j < Nf) {
            v = W[(size_t)k * Nf + j];
        } else if (H && j < Nf + 2 * H) {
            int hh = j - Nf;
            const float* a = as;
            if (hh >= H) { hh -= H; a = ad; }
            float s = 0.0f;
            for (int c = 0; c < C; ++c)
                s += W[(size_t)k * Nf + hh * C + c] * a[hh * C + c];
            v = s;
        }
    }
    out[((size_t)(k >> 5) * ldo + j) * 32 + aswz(k & 31)] = (_Float16)v;
}

// monotone float->uint key for atomic max over floats
__device__ __forceinline__ unsigned fkey(float f) {
    unsigned u = __float_as_uint(f);
    return (u & 0x80000000u) ? ~u : (u | 0x80000000u);
}
__device__ __forceinline__ float funkey(unsigned k) {
    unsigned u = (k & 0x80000000u) ? (k ^ 0x80000000u) : ~k;
    return __uint_as_float(u);
}

__global__ void init_softmax_kernel(unsigned* __restrict__ emax,
                                    float* __restrict__ denom, int n)
{
    int t = blockIdx.x * blockDim.x + threadIdx.x;
    if (t < n) { emax[t] = 0x007FFFFFu; /* key(-inf) */ denom[t] = 0.0f; }
}

// pass1: e = lrelu(logit_s[src] + logit_d[dst]); P=e; segment max into emax
__global__ void edge_pass1_kernel(const int* __restrict__ ei, int E, int Etot,
                                  const float* __restrict__ G, int ldg,
                                  int offs, int offd, int H,
                                  float* __restrict__ P,
                                  unsigned* __restrict__ emax)
{
    int t = blockIdx.x * blockDim.x + threadIdx.x;
    if (t >= Etot) return;
    int s, d;
    if (t < E) { s = ei[t]; d = ei[E + t]; } else { s = d = t - E; }
    for (int h = 0; h < H; ++h) {
        float e = G[(size_t)s * ldg + offs + h] + G[(size_t)d * ldg + offd + h];
        e = (e > 0.0f) ? e : NEG_SLOPE * e;
        P[(size_t)t * H + h] = e;
        atomicMax(&emax[d * H + h], fkey(e));
    }
}

// pass2: p = exp(e - emax[dst]); P=p; segment sum into denom
__global__ void edge_pass2_kernel(const int* __restrict__ ei, int E, int Etot,
                                  int H,
                                  float* __restrict__ P,
                                  const unsigned* __restrict__ emax,
                                  float* __restrict__ denom)
{
    int t = blockIdx.x * blockDim.x + threadIdx.x;
    if (t >= Etot) return;
    int d = (t < E) ? ei[E + t] : (t - E);
    for (int h = 0; h < H; ++h) {
        float m = funkey(emax[d * H + h]);
        float p = __expf(P[(size_t)t * H + h] - m);
        P[(size_t)t * H + h] = p;
        atomicAdd(&denom[d * H + h], p);
    }
}

// pass3: one wave per edge; AGG[dst, c] += G[src, c] * alpha[head(c)]
__global__ void edge_pass3_kernel(const int* __restrict__ ei, int E, int Etot,
                                  const float* __restrict__ G, int ldg,
                                  int H, int C,
                                  const float* __restrict__ P,
                                  const float* __restrict__ denom,
                                  float* __restrict__ AGG, int ldagg)
{
    int gw   = (blockIdx.x * blockDim.x + threadIdx.x) >> 5;
    int lane = threadIdx.x & 31;
    if (gw >= Etot) return;
    int s, d;
    if (gw < E) { s = ei[gw]; d = ei[E + gw]; } else { s = d = gw - E; }
    int HC = H * C;
    for (int c = lane; c < HC; c += 32) {
        int h = c / C;
        float den   = fmaxf(denom[d * H + h], 1e-16f);
        float alpha = P[(size_t)gw * H + h] / den;
        atomicAdd(&AGG[(size_t)d * ldagg + c],
                  G[(size_t)s * ldg + c] * alpha);
    }
}

// out_f16[M x ldp] (A-swizzled) = elu(AGG[M x n] + bias) (+ resid), pad cols = 0
__global__ void finalize_kernel(const float* __restrict__ AGG,
                                const float* __restrict__ bias,
                                const float* __restrict__ resid,
                                int n, int ldp, long long total,
                                _Float16* __restrict__ out)
{
    long long t = (long long)blockIdx.x * blockDim.x + threadIdx.x;
    if (t >= total) return;
    int j = (int)(t % ldp);
    long long r = t / ldp;
    float v = 0.0f;
    if (j < n) {
        v = AGG[r * n + j] + bias[j];
        v = (v > 0.0f) ? v : (__expf(v) - 1.0f);
        if (resid) v += resid[r * n + j];
    }
    out[r * ldp + (j & ~31) + aswz(j & 31)] = (_Float16)v;
}

// score[n] = bs2 + sum_j S1[n,j] * Ws2[j]   (j < 32)
__global__ void score_kernel(const float* __restrict__ S1,
                             const float* __restrict__ Ws2,
                             const float* __restrict__ bs2,
                             float* __restrict__ out, int M)
{
    int t = blockIdx.x * blockDim.x + threadIdx.x;
    if (t >= M) return;
    float acc = bs2[0];
    const float* row = S1 + (size_t)t * 32;
#pragma unroll
    for (int j = 0; j < 32; ++j) acc += row[j] * Ws2[j];
    out[t] = acc;
}

// ---------------------------------------------------------------------------
static void launch_gemm(const _Float16* A, int lda, const _Float16* B, int ldb,
                        float* C, int ldc, int M, int K, int Nact,
                        const float* bias, int act, hipStream_t s)
{
    dim3 grid(M / 16, (unsigned)(((Nact + 15) / 16 + 7) / 8));
    hipLaunchKernelGGL(wmma_gemm_kernel, grid, dim3(256), 0, s,
                       A, lda, B, ldb, C, ldc, K, Nact, bias, act);
}

extern "C" void kernel_launch(void* const* d_in, const int* in_sizes, int n_in,
                              void* d_out, int out_size, void* d_ws, size_t ws_size,
                              hipStream_t stream)
{
    const float* x    = (const float*)d_in[0];
    const int*   ei   = (const int*)  d_in[1];
    const float* W_in = (const float*)d_in[2];
    const float* b_in = (const float*)d_in[3];
    const float* W1   = (const float*)d_in[4];
    const float* as1  = (const float*)d_in[5];
    const float* ad1  = (const float*)d_in[6];
    const float* b1   = (const float*)d_in[7];
    const float* W2   = (const float*)d_in[8];
    const float* as2  = (const float*)d_in[9];
    const float* ad2  = (const float*)d_in[10];
    const float* b2   = (const float*)d_in[11];
    const float* W3   = (const float*)d_in[12];
    const float* as3  = (const float*)d_in[13];
    const float* ad3  = (const float*)d_in[14];
    const float* b3   = (const float*)d_in[15];
    const float* Ws1  = (const float*)d_in[16];
    const float* bs1  = (const float*)d_in[17];
    const float* Ws2  = (const float*)d_in[18];
    const float* bs2  = (const float*)d_in[19];
    float* out = (float*)d_out;

    const int IN   = 256;
    const int Nn   = in_sizes[0] / IN;       // 50000 (multiple of 16)
    const int E    = in_sizes[1] / 2;        // 800000
    const int Etot = E + Nn;                 // + self loops

    // ---- workspace carving (256B aligned) ----
    char* cur = (char*)d_ws;
    auto alloc = [&](size_t bytes) -> void* {
        void* p = (void*)cur;
        cur += (bytes + 255) & ~(size_t)255;
        return p;
    };
    _Float16* S    = (_Float16*)alloc((size_t)Nn * 288 * sizeof(_Float16)); // f16 staging
    float*    G    = (float*)   alloc((size_t)Nn * 304 * sizeof(float));    // gemm out + logits
    float*    AGG  = (float*)   alloc((size_t)Nn * 288 * sizeof(float));
    float*    x0   = (float*)   alloc((size_t)Nn * HID * sizeof(float));
    float*    P    = (float*)   alloc((size_t)Etot * HEADS * sizeof(float));
    unsigned* emax = (unsigned*)alloc((size_t)Nn * HEADS * sizeof(unsigned));
    float*    den  = (float*)   alloc((size_t)Nn * HEADS * sizeof(float));
    _Float16* Bin  = (_Float16*)alloc((size_t)256 * 80  * sizeof(_Float16));
    _Float16* B1   = (_Float16*)alloc((size_t)96  * 304 * sizeof(_Float16));
    _Float16* B2   = (_Float16*)alloc((size_t)288 * 80  * sizeof(_Float16));
    _Float16* B3   = (_Float16*)alloc((size_t)96  * 80  * sizeof(_Float16));
    _Float16* Bs1  = (_Float16*)alloc((size_t)96  * 32  * sizeof(_Float16));

    const int TPB = 256;
    auto blocks  = [&](long long n) { return (unsigned)((n + TPB - 1) / TPB); };

    // ---- weight prep (f16, packed layout, logit columns appended) ----
    hipLaunchKernelGGL(prep_b_kernel, dim3(blocks(256 * 80)), dim3(TPB), 0, stream,
                       W_in, nullptr, nullptr, 256, HID, 0, HID, Bin, 80, 256 * 80);
    hipLaunchKernelGGL(prep_b_kernel, dim3(blocks(96 * 304)), dim3(TPB), 0, stream,
                       W1, as1, ad1, HID, HEADS * HID, HEADS, HID, B1, 304, 96 * 304);
    hipLaunchKernelGGL(prep_b_kernel, dim3(blocks(288 * 80)), dim3(TPB), 0, stream,
                       W2, as2, ad2, HEADS * HID, HID, 1, HID, B2, 80, 288 * 80);
    hipLaunchKernelGGL(prep_b_kernel, dim3(blocks(96 * 80)), dim3(TPB), 0, stream,
                       W3, as3, ad3, HID, HID, 1, HID, B3, 80, 96 * 80);
    hipLaunchKernelGGL(prep_b_kernel, dim3(blocks(96 * 32)), dim3(TPB), 0, stream,
                       Ws1, nullptr, nullptr, HID, 32, 0, HID, Bs1, 32, 96 * 32);

    // ---- x0 = x @ W_in + b_in ----
    hipLaunchKernelGGL(convert_pad_kernel, dim3(blocks((long long)Nn * 256)), dim3(TPB), 0, stream,
                       x, S, 256, 256, (long long)Nn * 256);
    launch_gemm(S, 256, Bin, 80, x0, HID, Nn, 256, HID, b_in, 0, stream);

    // ===================== GAT layer 1 (4 heads, concat) =====================
    hipLaunchKernelGGL(convert_pad_kernel, dim3(blocks((long long)Nn * 96)), dim3(TPB), 0, stream,
                       x0, S, HID, 96, (long long)Nn * 96);
    launch_gemm(S, 96, B1, 304, G, 304, Nn, 96, 288 + 2 * HEADS, nullptr, 0, stream);

    hipLaunchKernelGGL(init_softmax_kernel, dim3(blocks(Nn * HEADS)), dim3(TPB), 0, stream,
                       emax, den, Nn * HEADS);
    hipMemsetAsync(AGG, 0, (size_t)Nn * 288 * sizeof(float), stream);
    hipLaunchKernelGGL(edge_pass1_kernel, dim3(blocks(Etot)), dim3(TPB), 0, stream,
                       ei, E, Etot, G, 304, 288, 292, HEADS, P, emax);
    hipLaunchKernelGGL(edge_pass2_kernel, dim3(blocks(Etot)), dim3(TPB), 0, stream,
                       ei, E, Etot, HEADS, P, emax, den);
    hipLaunchKernelGGL(edge_pass3_kernel, dim3((unsigned)((Etot + 7) / 8)), dim3(TPB), 0, stream,
                       ei, E, Etot, G, 304, HEADS, HID, P, den, AGG, 288);
    hipLaunchKernelGGL(finalize_kernel, dim3(blocks((long long)Nn * 288)), dim3(TPB), 0, stream,
                       AGG, b1, nullptr, 288, 288, (long long)Nn * 288, S);

    // ===================== GAT layer 2 (1 head) =====================
    launch_gemm(S, 288, B2, 80, G, 80, Nn, 288, HID + 2, nullptr, 0, stream);
    hipLaunchKernelGGL(init_softmax_kernel, dim3(blocks(Nn)), dim3(TPB), 0, stream,
                       emax, den, Nn);
    hipMemsetAsync(AGG, 0, (size_t)Nn * HID * sizeof(float), stream);
    hipLaunchKernelGGL(edge_pass1_kernel, dim3(blocks(Etot)), dim3(TPB), 0, stream,
                       ei, E, Etot, G, 80, 72, 73, 1, P, emax);
    hipLaunchKernelGGL(edge_pass2_kernel, dim3(blocks(Etot)), dim3(TPB), 0, stream,
                       ei, E, Etot, 1, P, emax, den);
    hipLaunchKernelGGL(edge_pass3_kernel, dim3((unsigned)((Etot + 7) / 8)), dim3(TPB), 0, stream,
                       ei, E, Etot, G, 80, 1, HID, P, den, AGG, HID);
    hipLaunchKernelGGL(finalize_kernel, dim3(blocks((long long)Nn * 96)), dim3(TPB), 0, stream,
                       AGG, b2, nullptr, HID, 96, (long long)Nn * 96, S);

    // ===================== GAT layer 3 (1 head) + residual =====================
    launch_gemm(S, 96, B3, 80, G, 80, Nn, 96, HID + 2, nullptr, 0, stream);
    hipLaunchKernelGGL(init_softmax_kernel, dim3(blocks(Nn)), dim3(TPB), 0, stream,
                       emax, den, Nn);
    hipMemsetAsync(AGG, 0, (size_t)Nn * HID * sizeof(float), stream);
    hipLaunchKernelGGL(edge_pass1_kernel, dim3(blocks(Etot)), dim3(TPB), 0, stream,
                       ei, E, Etot, G, 80, 72, 73, 1, P, emax);
    hipLaunchKernelGGL(edge_pass2_kernel, dim3(blocks(Etot)), dim3(TPB), 0, stream,
                       ei, E, Etot, 1, P, emax, den);
    hipLaunchKernelGGL(edge_pass3_kernel, dim3((unsigned)((Etot + 7) / 8)), dim3(TPB), 0, stream,
                       ei, E, Etot, G, 80, 1, HID, P, den, AGG, HID);
    // h = elu(agg + b3) + x0, staged (swizzled f16) for scoring GEMM
    hipLaunchKernelGGL(finalize_kernel, dim3(blocks((long long)Nn * 96)), dim3(TPB), 0, stream,
                       AGG, b3, x0, HID, 96, (long long)Nn * 96, S);

    // ===================== scoring head =====================
    launch_gemm(S, 96, Bs1, 32, G, 32, Nn, 96, 32, bs1, 2, stream);   // relu(h@Ws1+bs1)
    hipLaunchKernelGGL(score_kernel, dim3(blocks(Nn)), dim3(TPB), 0, stream,
                       G, Ws2, bs2, out, Nn);
}
